// MultiHeadAttention_45586782880331
// MI455X (gfx1250) — matrix operand order repaired and checked by reading, not compiled, not run on hardware
//
#include <hip/hip_runtime.h>
#include <stdint.h>

// ---------------------------------------------------------------- types
typedef __bf16 bf16;
typedef __attribute__((ext_vector_type(16))) __bf16 v16bf;
typedef __attribute__((ext_vector_type(8)))  __bf16 bf16x8;
typedef __attribute__((ext_vector_type(4)))  __bf16 bf16x4;
typedef __attribute__((ext_vector_type(8)))  float  v8f;

#define NB_B 2
#define NB_S 2048
#define DMODEL 1024
#define NB_H 16
#define HEADD 64
#define LN_EPS 1e-5f

// ---------------------------------------------------------------- CDNA5 async copy (global -> LDS, ASYNCcnt)
// LDS destination = byte address held in a VGPR; vaddr = 64-bit global address (GV mode).
__device__ __forceinline__ void async_b128(uint32_t lds_byte_addr, const void* gptr) {
  asm volatile("global_load_async_to_lds_b128 %0, %1, off"
               :: "v"(lds_byte_addr), "v"(gptr)
               : "memory");
}
__device__ __forceinline__ void wait_async0() {
  asm volatile("s_wait_asynccnt 0x0" ::: "memory");
}
__device__ __forceinline__ uint32_t lds_addr(const void* p) {
  return (uint32_t)(uintptr_t)p;   // LDS aperture: addr[31:0] is the LDS byte offset
}

// ---------------------------------------------------------------- fragment loaders
// A fragment (M=16, K=32, 16-bit): lanes 0-15 hold row M=lane, K {0..7,16..23};
// lanes 16-31 hold row M=lane-16, K {8..15, 24..31}.  (ISA 7.12.2)
__device__ __forceinline__ v16bf ld_frag_a(const bf16* __restrict__ base, int ld, int lane) {
  const int r  = lane & 15;
  const int hi = lane >> 4;
  const bf16* p = base + (size_t)r * ld + hi * 8;
  bf16x8 lo  = *(const bf16x8*)(p);        // K = hi*8 .. hi*8+7
  bf16x8 hi8 = *(const bf16x8*)(p + 16);   // K = 16 + hi*8 .. +7
  return __builtin_shufflevector(lo, hi8, 0,1,2,3,4,5,6,7,8,9,10,11,12,13,14,15);
}

// B fragment (K=32, N=16, 16-bit), stored as [N rows][K contiguous]:
// lanes 0-15 hold col N=lane with K {0..7,16..23}; lanes 16-31 hold K {8..15,24..31}.
__device__ __forceinline__ v16bf ld_frag_b(const bf16* __restrict__ base, int ld, int lane) {
  const int r  = lane & 15;
  const int hi = lane >> 4;
  const bf16* p = base + (size_t)r * ld + hi * 8;
  bf16x8 lo  = *(const bf16x8*)(p);
  bf16x8 hi8 = *(const bf16x8*)(p + 16);
  return __builtin_shufflevector(lo, hi8, 0,1,2,3,4,5,6,7,8,9,10,11,12,13,14,15);
}

// ---------------------------------------------------------------- fp32 -> bf16 convert (vectorized)
__global__ __launch_bounds__(256) void cvt_f32_bf16(const float4* __restrict__ src,
                                                    bf16x4* __restrict__ dst, int n4) {
  for (int i = blockIdx.x * blockDim.x + threadIdx.x; i < n4; i += gridDim.x * blockDim.x) {
    float4 f = src[i];
    bf16x4 o = {(bf16)f.x, (bf16)f.y, (bf16)f.z, (bf16)f.w};
    dst[i] = o;
  }
}

// ---------------------------------------------------------------- WMMA GEMM  C = A * W^T + bias
// A: [M,K] bf16 row-major, W: [N,K] bf16 row-major. Tile 128x128x32, 8 waves (4M x 2N),
// each wave 32x64 = 2x4 C-fragments. Double-buffered async LDS staging.
template <bool BF_OUT>
__global__ __launch_bounds__(256)
void gemm_bf16(const bf16* __restrict__ A, const bf16* __restrict__ W,
               const float* __restrict__ bias, const float* __restrict__ resid,
               bf16* __restrict__ outb, float* __restrict__ outf,
               float scale, int M, int N, int K) {
  constexpr int LDT = 40;                       // 32 + 8 pad (keeps 16B alignment)
  __shared__ bf16 As[2][128 * LDT];
  __shared__ bf16 Bs[2][128 * LDT];

  const int t    = threadIdx.x;
  const int lane = t & 31;
  const int wave = t >> 5;
  const int m0 = blockIdx.y * 128;
  const int n0 = blockIdx.x * 128;
  const int wm = (wave >> 1) * 32;              // wave row offset in tile
  const int wn = (wave & 1)  * 64;              // wave col offset in tile

  v8f zero = {};
  v8f acc[2][4];
  for (int i = 0; i < 2; i++)
    for (int j = 0; j < 4; j++) acc[i][j] = zero;

  const int ldr = t >> 1;                       // staging row 0..127
  const int ldc = (t & 1) * 16;                 // staging col 0 or 16
  const uint32_t la[2] = {lds_addr(&As[0][ldr * LDT + ldc]), lds_addr(&As[1][ldr * LDT + ldc])};
  const uint32_t lb[2] = {lds_addr(&Bs[0][ldr * LDT + ldc]), lds_addr(&Bs[1][ldr * LDT + ldc])};
  const bf16* gA = A + (size_t)(m0 + ldr) * K + ldc;
  const bf16* gW = W + (size_t)(n0 + ldr) * K + ldc;

  auto issue = [&](int k0, int buf) {
    async_b128(la[buf],      gA + k0);
    async_b128(la[buf] + 16, gA + k0 + 8);
    async_b128(lb[buf],      gW + k0);
    async_b128(lb[buf] + 16, gW + k0 + 8);
  };

  issue(0, 0);
  const int nk = K / 32;
  for (int i = 0; i < nk; i++) {
    const int cur = i & 1;
    wait_async0();                              // own async copies done
    __syncthreads();                            // everyone's copies visible
    if (i + 1 < nk) issue((i + 1) * 32, cur ^ 1);

    v16bf af[2], bfr[4];
    for (int mt = 0; mt < 2; mt++) af[mt]  = ld_frag_a(&As[cur][(wm + mt * 16) * LDT], LDT, lane);
    for (int nt = 0; nt < 4; nt++) bfr[nt] = ld_frag_b(&Bs[cur][(wn + nt * 16) * LDT], LDT, lane);
    for (int mt = 0; mt < 2; mt++)
      for (int nt = 0; nt < 4; nt++)
        acc[mt][nt] = __builtin_amdgcn_wmma_f32_16x16x32_bf16(
            false, af[mt], false, bfr[nt], (short)0, acc[mt][nt], false, false);
    __syncthreads();                            // reads done before buffer reuse
  }

  // C layout: col = lane&15, row = vgpr + 8*(lane>=16)   (ISA 7.12.2)
  const int nl = lane & 15;
  const int mh = (lane >> 4) * 8;
  for (int mt = 0; mt < 2; mt++)
    for (int nt = 0; nt < 4; nt++)
      for (int v = 0; v < 8; v++) {
        const int row = m0 + wm + mt * 16 + mh + v;
        const int col = n0 + wn + nt * 16 + nl;
        float val = acc[mt][nt][v] + bias[col];
        if constexpr (BF_OUT) {
          outb[(size_t)row * N + col] = (bf16)(val * scale);
        } else {
          outf[(size_t)row * N + col] = val + resid[(size_t)row * N + col];
        }
      }
}

// ---------------------------------------------------------------- flash attention
// One block = (b, h, 128 query rows); each wave owns 16 query rows.
// Q pre-scaled by 1/sqrt(HD). K tile async-copied to LDS; V tile transposed into LDS.
__global__ __launch_bounds__(256)
void flash_attn(const bf16* __restrict__ Q, const bf16* __restrict__ K,
                const bf16* __restrict__ V, bf16* __restrict__ O) {
  constexpr int LDK = 72, LDVT = 72, LDP = 72;  // 64 + 8 pad
  __shared__ bf16 Ks [64 * LDK];                // [key][d]
  __shared__ bf16 Vst[64 * LDVT];               // [d][key] (transposed)
  __shared__ bf16 Ps [8 * 16 * LDP];            // per-wave P scratch [16 rows][64 keys]

  const int t    = threadIdx.x;
  const int lane = t & 31;
  const int w    = t >> 5;
  const int h    = blockIdx.y;
  const int b    = blockIdx.z;
  const int q0   = blockIdx.x * 128 + w * 16;
  const size_t headoff = (size_t)h * HEADD;
  const size_t rowbase = (size_t)b * NB_S * DMODEL;

  v16bf aq0 = ld_frag_a(Q + rowbase + (size_t)q0 * DMODEL + headoff,      DMODEL, lane);
  v16bf aq1 = ld_frag_a(Q + rowbase + (size_t)q0 * DMODEL + headoff + 32, DMODEL, lane);

  v8f zero = {};
  v8f acc[4] = {zero, zero, zero, zero};
  float mrun[8], lrun[8];
  for (int v = 0; v < 8; v++) { mrun[v] = -1e30f; lrun[v] = 0.0f; }

  const int ldr = t >> 2;                       // key row 0..63
  const int ldc = (t & 3) * 16;                 // d chunk
  const int nl  = lane & 15;
  const int mh  = (lane >> 4) * 8;
  const uint32_t lk = lds_addr(&Ks[ldr * LDK + ldc]);
  const bf16* gK = K + rowbase + headoff + (size_t)ldr * DMODEL + ldc;
  const bf16* gV = V + rowbase + headoff + (size_t)ldr * DMODEL + ldc;

  for (int kt = 0; kt < NB_S / 64; kt++) {
    __syncthreads();                            // previous tile fully consumed
    {
      const size_t koff = (size_t)kt * 64 * DMODEL;
      async_b128(lk,      gK + koff);           // K tile: straight async copy
      async_b128(lk + 16, gK + koff + 8);
      bf16x8 v0 = *(const bf16x8*)(gV + koff);  // V tile: transpose via regs
      bf16x8 v1 = *(const bf16x8*)(gV + koff + 8);
      for (int j = 0; j < 8; j++) {
        Vst[(ldc + j) * LDVT + ldr]     = v0[j];
        Vst[(ldc + 8 + j) * LDVT + ldr] = v1[j];
      }
    }
    wait_async0();
    __syncthreads();

    // scores: S = Q * K^T  (Q pre-scaled)
    v8f sc[4];
    for (int nt = 0; nt < 4; nt++) {
      v8f s = zero;
      v16bf bk0 = ld_frag_b(&Ks[nt * 16 * LDK],      LDK, lane);
      v16bf bk1 = ld_frag_b(&Ks[nt * 16 * LDK + 32], LDK, lane);
      s = __builtin_amdgcn_wmma_f32_16x16x32_bf16(false, aq0, false, bk0, (short)0, s, false, false);
      s = __builtin_amdgcn_wmma_f32_16x16x32_bf16(false, aq1, false, bk1, (short)0, s, false, false);
      sc[nt] = s;
    }

    // online softmax for this lane's 8 rows (rows mh..mh+7, reduce across 16-lane half)
    float mnew[8], alpha[8];
    for (int v = 0; v < 8; v++) {
      float tm = fmaxf(fmaxf(sc[0][v], sc[1][v]), fmaxf(sc[2][v], sc[3][v]));
      tm = fmaxf(tm, __shfl_xor(tm, 1, 32));
      tm = fmaxf(tm, __shfl_xor(tm, 2, 32));
      tm = fmaxf(tm, __shfl_xor(tm, 4, 32));
      tm = fmaxf(tm, __shfl_xor(tm, 8, 32));
      mnew[v]  = fmaxf(mrun[v], tm);
      alpha[v] = __expf(mrun[v] - mnew[v]);
      mrun[v]  = mnew[v];
    }
    for (int v = 0; v < 8; v++) {
      float sum = 0.0f;
      for (int nt = 0; nt < 4; nt++) {
        float p = __expf(sc[nt][v] - mnew[v]);
        Ps[(w * 16 + mh + v) * LDP + nt * 16 + nl] = (bf16)p;
        sum += p;
      }
      sum += __shfl_xor(sum, 1, 32);
      sum += __shfl_xor(sum, 2, 32);
      sum += __shfl_xor(sum, 4, 32);
      sum += __shfl_xor(sum, 8, 32);
      lrun[v] = lrun[v] * alpha[v] + sum;
      for (int nt = 0; nt < 4; nt++) acc[nt][v] *= alpha[v];
    }

    // ctx += P * V
    v16bf ap0 = ld_frag_a(&Ps[w * 16 * LDP],      LDP, lane);
    v16bf ap1 = ld_frag_a(&Ps[w * 16 * LDP + 32], LDP, lane);
    for (int nt = 0; nt < 4; nt++) {
      v16bf bv0 = ld_frag_b(&Vst[nt * 16 * LDVT],      LDVT, lane);
      v16bf bv1 = ld_frag_b(&Vst[nt * 16 * LDVT + 32], LDVT, lane);
      acc[nt] = __builtin_amdgcn_wmma_f32_16x16x32_bf16(false, ap0, false, bv0, (short)0, acc[nt], false, false);
      acc[nt] = __builtin_amdgcn_wmma_f32_16x16x32_bf16(false, ap1, false, bv1, (short)0, acc[nt], false, false);
    }
  }

  for (int nt = 0; nt < 4; nt++)
    for (int v = 0; v < 8; v++) {
      float o = acc[nt][v] / lrun[v];
      O[rowbase + (size_t)(q0 + mh + v) * DMODEL + headoff + nt * 16 + nl] = (bf16)o;
    }
}

// ---------------------------------------------------------------- LayerNorm (one block per row)
__global__ __launch_bounds__(256)
void layernorm(const float* __restrict__ y, const float* __restrict__ gamma,
               const float* __restrict__ beta, float* __restrict__ out) {
  __shared__ float red[16];
  const int row = blockIdx.x;
  const int t = threadIdx.x;
  const float* yr = y + (size_t)row * DMODEL;
  float4 vv = *(const float4*)(yr + t * 4);
  float s  = vv.x + vv.y + vv.z + vv.w;
  float s2 = vv.x * vv.x + vv.y * vv.y + vv.z * vv.z + vv.w * vv.w;
  for (int m = 1; m < 32; m <<= 1) { s += __shfl_xor(s, m, 32); s2 += __shfl_xor(s2, m, 32); }
  const int lane = t & 31, w = t >> 5;
  if (lane == 0) { red[w] = s; red[8 + w] = s2; }
  __syncthreads();
  float ts = 0.0f, ts2 = 0.0f;
  for (int i = 0; i < 8; i++) { ts += red[i]; ts2 += red[8 + i]; }
  const float mu   = ts / DMODEL;
  const float var  = ts2 / DMODEL - mu * mu;
  const float rstd = rsqrtf(var + LN_EPS);
  float4 g  = *(const float4*)(gamma + t * 4);
  float4 be = *(const float4*)(beta + t * 4);
  float4 o;
  o.x = (vv.x - mu) * rstd * g.x + be.x;
  o.y = (vv.y - mu) * rstd * g.y + be.y;
  o.z = (vv.z - mu) * rstd * g.z + be.z;
  o.w = (vv.w - mu) * rstd * g.w + be.w;
  *(float4*)(out + (size_t)row * DMODEL + t * 4) = o;
}

// ---------------------------------------------------------------- launch
extern "C" void kernel_launch(void* const* d_in, const int* in_sizes, int n_in,
                              void* d_out, int out_size, void* d_ws, size_t ws_size,
                              hipStream_t stream) {
  (void)in_sizes; (void)n_in; (void)out_size; (void)ws_size;
  const float* x     = (const float*)d_in[0];
  const float* Wq    = (const float*)d_in[1];
  const float* bq    = (const float*)d_in[2];
  const float* Wk    = (const float*)d_in[3];
  const float* bk    = (const float*)d_in[4];
  const float* Wv    = (const float*)d_in[5];
  const float* bv    = (const float*)d_in[6];
  const float* Wo    = (const float*)d_in[7];
  const float* bo    = (const float*)d_in[8];
  const float* gamma = (const float*)d_in[9];
  const float* beta  = (const float*)d_in[10];
  float* out = (float*)d_out;

  const int    MT = NB_B * NB_S;                // 4096 tokens
  const size_t EL = (size_t)MT * DMODEL;        // 4M elements
  const size_t WE = (size_t)DMODEL * DMODEL;    // 1M elements

  char* ws = (char*)d_ws;
  bf16* xbf  = (bf16*)ws;  ws += EL * 2;
  bf16* wqbf = (bf16*)ws;  ws += WE * 2;
  bf16* wkbf = (bf16*)ws;  ws += WE * 2;
  bf16* wvbf = (bf16*)ws;  ws += WE * 2;
  bf16* wobf = (bf16*)ws;  ws += WE * 2;
  bf16* Qbf  = (bf16*)ws;  ws += EL * 2;
  bf16* Kbf  = (bf16*)ws;  ws += EL * 2;
  bf16* Vbf  = (bf16*)ws;  ws += EL * 2;
  bf16* Cbf  = (bf16*)ws;  ws += EL * 2;
  float* yb  = (float*)ws; ws += EL * 4;

  cvt_f32_bf16<<<1024, 256, 0, stream>>>((const float4*)x,  (bf16x4*)xbf,  (int)(EL / 4));
  cvt_f32_bf16<<<256,  256, 0, stream>>>((const float4*)Wq, (bf16x4*)wqbf, (int)(WE / 4));
  cvt_f32_bf16<<<256,  256, 0, stream>>>((const float4*)Wk, (bf16x4*)wkbf, (int)(WE / 4));
  cvt_f32_bf16<<<256,  256, 0, stream>>>((const float4*)Wv, (bf16x4*)wvbf, (int)(WE / 4));
  cvt_f32_bf16<<<256,  256, 0, stream>>>((const float4*)Wo, (bf16x4*)wobf, (int)(WE / 4));

  dim3 gg(DMODEL / 128, MT / 128);              // (8, 32)
  const float qscale = 0.125f;                  // 1/sqrt(64)
  gemm_bf16<true ><<<gg, 256, 0, stream>>>(xbf, wqbf, bq, nullptr, Qbf, nullptr, qscale, MT, DMODEL, DMODEL);
  gemm_bf16<true ><<<gg, 256, 0, stream>>>(xbf, wkbf, bk, nullptr, Kbf, nullptr, 1.0f,   MT, DMODEL, DMODEL);
  gemm_bf16<true ><<<gg, 256, 0, stream>>>(xbf, wvbf, bv, nullptr, Vbf, nullptr, 1.0f,   MT, DMODEL, DMODEL);

  flash_attn<<<dim3(NB_S / 128, NB_H, NB_B), 256, 0, stream>>>(Qbf, Kbf, Vbf, Cbf);

  gemm_bf16<false><<<gg, 256, 0, stream>>>(Cbf, wobf, bo, x, nullptr, yb, 1.0f, MT, DMODEL, DMODEL);

  layernorm<<<MT, 256, 0, stream>>>(yb, gamma, beta, out);
}
